// Block_21242908246745
// MI455X (gfx1250) — matrix operand order repaired and checked by reading, not compiled, not run on hardware
//
#include <hip/hip_runtime.h>
#include <hip/hip_bf16.h>

// ---------------------------------------------------------------------------
// Problem constants
// ---------------------------------------------------------------------------
#define B_SZ    32
#define N_SEQ   577
#define D_MODEL 384
#define N_HEADS 6
#define HD      64
#define HIDDEN  1536
#define ROWS    (B_SZ * N_SEQ)                       /* 18464 */
#define QKV_STRIDE (B_SZ * N_HEADS * N_SEQ * HD)     /* 7090176 */

typedef __attribute__((ext_vector_type(16))) __bf16 v16bf;
typedef __attribute__((ext_vector_type(8)))  float  v8f;

union Frag {
  v16bf v;
  unsigned int   u32[8];
  unsigned short u16[16];
};

__device__ __forceinline__ unsigned short f2bf(float f) {
  unsigned int u = __float_as_uint(f);
  u += 0x7FFFu + ((u >> 16) & 1u);      // round-to-nearest-even
  return (unsigned short)(u >> 16);
}

__device__ __forceinline__ v8f wmma_bf16(const Frag& a, const Frag& b, v8f c) {
  return __builtin_amdgcn_wmma_f32_16x16x32_bf16(
      /*neg_a=*/false, a.v, /*neg_b=*/false, b.v,
      /*c_mod=*/(short)0, c, /*reuse_a=*/false, /*reuse_b=*/false);
}

// ---------------------------------------------------------------------------
// f32 -> bf16 weight conversion
// ---------------------------------------------------------------------------
__global__ void cvt_f32_bf16(const float* __restrict__ in,
                             unsigned short* __restrict__ out, int n) {
  int i = blockIdx.x * blockDim.x + threadIdx.x;
  if (i < n) out[i] = f2bf(in[i]);
}

// ---------------------------------------------------------------------------
// LayerNorm over D=384, one block (128 threads) per row, bf16 output
// ---------------------------------------------------------------------------
__global__ __launch_bounds__(128) void ln_kernel(const float* __restrict__ x,
                                                 const float* __restrict__ g,
                                                 const float* __restrict__ b,
                                                 unsigned short* __restrict__ out) {
  int row = blockIdx.x, t = threadIdx.x;
  __shared__ float red[128];
  const float* xr = x + (size_t)row * D_MODEL;
  float v0 = xr[t], v1 = xr[t + 128], v2 = xr[t + 256];
  red[t] = v0 + v1 + v2;
  __syncthreads();
  for (int off = 64; off > 0; off >>= 1) {
    if (t < off) red[t] += red[t + off];
    __syncthreads();
  }
  float mu = red[0] * (1.0f / D_MODEL);
  __syncthreads();
  float d0 = v0 - mu, d1 = v1 - mu, d2 = v2 - mu;
  red[t] = d0 * d0 + d1 * d1 + d2 * d2;
  __syncthreads();
  for (int off = 64; off > 0; off >>= 1) {
    if (t < off) red[t] += red[t + off];
    __syncthreads();
  }
  float rstd = rsqrtf(red[0] * (1.0f / D_MODEL) + 1e-5f);
  unsigned short* orow = out + (size_t)row * D_MODEL;
  orow[t]       = f2bf(d0 * rstd * g[t]       + b[t]);
  orow[t + 128] = f2bf(d1 * rstd * g[t + 128] + b[t + 128]);
  orow[t + 256] = f2bf(d2 * rstd * g[t + 256] + b[t + 256]);
}

// ---------------------------------------------------------------------------
// bf16 WMMA GEMM:  C[M,N] = A[M,K] * W[K,N] + bias  (+ templated epilogue)
//   EPI 0: scatter to q/k/v [B,H,N,hd] (bf16)
//   EPI 1: + resid(x)  -> f32 (x2)
//   EPI 2: PolyGELU    -> bf16 (h3)
//   EPI 3: + resid(x2) -> f32 (d_out)
// Workgroup: 256 threads (8 waves as 4 row-waves x 2 col-waves).
// Tile 128 rows x 128 cols, K-step 32; 8 WMMAs per wave per K-step.
// W tile staged pair-interleaved so B-fragments load as aligned b32.
// ---------------------------------------------------------------------------
#define BP_STRIDE 258   /* (128 cols * 2) + 2 pad, in ushorts, per K-pair row */

template <int EPI>
__global__ __launch_bounds__(256) void gemm_kernel(
    const unsigned short* __restrict__ A, const unsigned short* __restrict__ W,
    const float* __restrict__ bias, int M, int N, int K,
    const float* __restrict__ resid, float* __restrict__ outf,
    unsigned short* __restrict__ outb, const float* __restrict__ gabc) {
  __shared__ alignas(16) unsigned short a_lds[128 * 34];       // 8704 B
  __shared__ alignas(16) unsigned short bp_lds[16 * BP_STRIDE]; // 8256 B

  const int tid  = threadIdx.x;
  const int wv   = tid >> 5;
  const int lane = tid & 31;
  const int wr   = wv & 3;                 // row-wave 0..3  (32 rows each)
  const int wc   = wv >> 2;                // col-wave 0..1  (64 cols each)
  const int m    = lane & 15;
  const int kbA  = (lane < 16) ? 0 : 8;    // A-frag K base per lane-half
  const int kpB  = (lane < 16) ? 0 : 4;    // B-frag K-pair base per lane-half
  const int rowBase = blockIdx.y * 128;
  const int colBase = blockIdx.x * 128;

  v8f acc[2][4];
#pragma unroll
  for (int h = 0; h < 2; ++h)
#pragma unroll
    for (int t = 0; t < 4; ++t) acc[h][t] = (v8f){0, 0, 0, 0, 0, 0, 0, 0};

  for (int k0 = 0; k0 < K; k0 += 32) {
    __syncthreads();
    // --- stage A tile: 128 x 32 bf16 (row-major, padded stride 34) ---
#pragma unroll
    for (int ii = 0; ii < 8; ++ii) {
      int d = tid + 256 * ii;
      int r = d >> 4, c = (d & 15) * 2;
      int gr = rowBase + r;
      if (gr > M - 1) gr = M - 1;
      const unsigned short* src = &A[(size_t)gr * K + k0 + c];
      *(unsigned int*)&a_lds[r * 34 + c] = *(const unsigned int*)src;
      if (k0 + 32 < K) __builtin_prefetch(src + 32, 0, 0);  // global_prefetch_b8
    }
    // --- stage W tile: 32 x 128 bf16, pair-interleaved: (k,n) ->
    //     bp_lds[(k>>1)*BP_STRIDE + 2n + (k&1)]  (2x2 repack per thread) ---
#pragma unroll
    for (int ii = 0; ii < 4; ++ii) {
      int bid = tid + 256 * ii;            // 1024 2x2 blocks
      int kp = bid >> 6, np = bid & 63;    // kp: 0..15, np: 0..63
      int k = 2 * kp, n = 2 * np;
      unsigned int a0 = *(const unsigned int*)&W[(size_t)(k0 + k) * N + colBase + n];
      unsigned int a1 = *(const unsigned int*)&W[(size_t)(k0 + k + 1) * N + colBase + n];
      *(unsigned int*)&bp_lds[kp * BP_STRIDE + n * 2]     = (a0 & 0xFFFFu) | (a1 << 16);
      *(unsigned int*)&bp_lds[kp * BP_STRIDE + n * 2 + 2] = (a0 >> 16) | (a1 & 0xFFFF0000u);
    }
    __syncthreads();

    // Two A fragments (rows wr*32 .. wr*32+31)
    Frag af[2];
#pragma unroll
    for (int h = 0; h < 2; ++h) {
#pragma unroll
      for (int i = 0; i < 8; ++i) {
        int k = kbA + (i >> 2) * 16 + (i & 3) * 2;
        af[h].u32[i] = *(const unsigned int*)&a_lds[(wr * 32 + h * 16 + m) * 34 + k];
      }
    }
    // 4 B fragments (cols wc*64 + t*16), each register one aligned b32 load
#pragma unroll
    for (int t = 0; t < 4; ++t) {
      Frag bfr;
      int col = wc * 64 + t * 16 + m;
#pragma unroll
      for (int i = 0; i < 8; ++i)
        bfr.u32[i] = *(const unsigned int*)&bp_lds[(kpB + (i >> 2) * 8 + (i & 3)) * BP_STRIDE + col * 2];
      acc[0][t] = wmma_bf16(af[0], bfr, acc[0][t]);
      acc[1][t] = wmma_bf16(af[1], bfr, acc[1][t]);
    }
  }

  // --- epilogue ---
  const int hi8 = (lane < 16) ? 0 : 8;
#pragma unroll
  for (int h = 0; h < 2; ++h) {
#pragma unroll
    for (int t = 0; t < 4; ++t) {
#pragma unroll
      for (int i = 0; i < 8; ++i) {
        int grow = rowBase + wr * 32 + h * 16 + i + hi8;
        int gcol = colBase + wc * 64 + t * 16 + m;
        if (grow >= M) continue;
        float val = acc[h][t][i] + bias[gcol];
        if (EPI == 0) {  // q/k/v scatter
          int which = gcol / D_MODEL;
          int rem   = gcol - which * D_MODEL;
          int head  = rem >> 6, hd = rem & 63;
          int bb = grow / N_SEQ, n = grow - bb * N_SEQ;
          outb[(size_t)which * QKV_STRIDE +
               (((size_t)bb * N_HEADS + head) * N_SEQ + n) * HD + hd] = f2bf(val);
        } else if (EPI == 1) {  // proj + residual -> f32
          outf[(size_t)grow * D_MODEL + gcol] =
              val + resid[(size_t)grow * D_MODEL + gcol];
        } else if (EPI == 2) {  // PolyGELU -> bf16
          float ga = gabc[0], gb = gabc[1], gc = gabc[2];
          outb[(size_t)grow * HIDDEN + gcol] = f2bf(ga * val * val + gb * val + gc);
        } else {  // fc2 + residual -> f32 output
          outf[(size_t)grow * D_MODEL + gcol] =
              val + resid[(size_t)grow * D_MODEL + gcol];
        }
      }
    }
  }
}

// ---------------------------------------------------------------------------
// Polynomial attention, one workgroup per (b,h).
// Streaming over 32-key blocks (poly >= 0, no softmax max-trick needed).
// K tile row-major (score B-frags already contiguous); V tile pair-interleaved
// so attn@v B-frags load as aligned b32.
// ---------------------------------------------------------------------------
#define VP_STRIDE 130   /* (64 cols * 2) + 2 pad, in ushorts, per K-pair row */

__global__ __launch_bounds__(256) void attn_kernel(
    const unsigned short* __restrict__ qg, const unsigned short* __restrict__ kg,
    const unsigned short* __restrict__ vg, const float* __restrict__ abc,
    unsigned short* __restrict__ out) {
  __shared__ alignas(16) unsigned short k_lds[32 * 66];          // 4224 B
  __shared__ alignas(16) unsigned short vp_lds[16 * VP_STRIDE];  // 4160 B
  __shared__ alignas(16) unsigned short p_lds[8 * 16 * 34];      // 8704 B

  const int bh = blockIdx.x;
  const int b = bh / N_HEADS, h = bh - b * N_HEADS;
  const int tid = threadIdx.x, wv = tid >> 5, lane = tid & 31;
  const int m   = lane & 15;
  const int kbA = (lane < 16) ? 0 : 8;
  const int kbB = (lane < 16) ? 0 : 16;
  const int kpB = (lane < 16) ? 0 : 4;
  const int hi8 = (lane < 16) ? 0 : 8;
  const float ca = abc[0], cb = abc[1], cc = abc[2];

  const unsigned short* qh = qg + (size_t)bh * N_SEQ * HD;
  const unsigned short* kh = kg + (size_t)bh * N_SEQ * HD;
  const unsigned short* vh = vg + (size_t)bh * N_SEQ * HD;
  unsigned short* pw = p_lds + wv * 544;  // 16 x 34

  const int NRT = 37;  // ceil(577/16) row tiles
  const int NJB = 19;  // ceil(577/32) key blocks

  for (int r = 0; r < 5; ++r) {
    int rt = r * 8 + wv;
    bool active = rt < NRT;
    int rtc = active ? rt : NRT - 1;

    // Q fragments for this row tile: 16x64 -> two 16x32 A-frags
    Frag qa0, qa1;
    {
      int row = rtc * 16 + m;
      if (row > N_SEQ - 1) row = N_SEQ - 1;
      const unsigned short* qr = qh + (size_t)row * HD;
#pragma unroll
      for (int i = 0; i < 8; ++i) {
        int k = kbA + (i >> 2) * 16 + (i & 3) * 2;
        qa0.u32[i] = *(const unsigned int*)&qr[k];
        qa1.u32[i] = *(const unsigned int*)&qr[k + 32];
      }
    }

    v8f oacc[4];
#pragma unroll
    for (int t = 0; t < 4; ++t) oacc[t] = (v8f){0, 0, 0, 0, 0, 0, 0, 0};
    float srow[8];
#pragma unroll
    for (int i = 0; i < 8; ++i) srow[i] = 0.f;

    for (int jb = 0; jb < NJB; ++jb) {
      int jj = jb * 32;
      __syncthreads();
      // stage K tile row-major (32 keys x 64 features)
#pragma unroll
      for (int ii = 0; ii < 4; ++ii) {
        int d = tid + 256 * ii;
        int kr = d >> 5, c = (d & 31) * 2;
        int gr = jj + kr;
        if (gr > N_SEQ - 1) gr = N_SEQ - 1;
        *(unsigned int*)&k_lds[kr * 66 + c] =
            *(const unsigned int*)&kh[(size_t)gr * HD + c];
      }
      // stage V tile pair-interleaved: (k,n) -> vp_lds[(k>>1)*VP_STRIDE+2n+(k&1)]
#pragma unroll
      for (int ii = 0; ii < 2; ++ii) {
        int bid = tid + 256 * ii;            // 512 2x2 blocks
        int kp = bid >> 5, np = bid & 31;    // kp: 0..15, np: 0..31
        int k = 2 * kp, n = 2 * np;
        int g0 = jj + k;     if (g0 > N_SEQ - 1) g0 = N_SEQ - 1;
        int g1 = jj + k + 1; if (g1 > N_SEQ - 1) g1 = N_SEQ - 1;
        unsigned int a0 = *(const unsigned int*)&vh[(size_t)g0 * HD + n];
        unsigned int a1 = *(const unsigned int*)&vh[(size_t)g1 * HD + n];
        *(unsigned int*)&vp_lds[kp * VP_STRIDE + n * 2]     = (a0 & 0xFFFFu) | (a1 << 16);
        *(unsigned int*)&vp_lds[kp * VP_STRIDE + n * 2 + 2] = (a0 >> 16) | (a1 & 0xFFFF0000u);
      }
      __syncthreads();

      // scores = q @ k^T  for 2 key tiles of 16, K=64 via two k=32 WMMAs each
      v8f s0 = (v8f){0, 0, 0, 0, 0, 0, 0, 0};
      v8f s1 = (v8f){0, 0, 0, 0, 0, 0, 0, 0};
#pragma unroll
      for (int half = 0; half < 2; ++half) {
        Frag kf0, kf1;
#pragma unroll
        for (int i = 0; i < 8; ++i) {
          int f = half * 32 + kbA + (i >> 2) * 16 + (i & 3) * 2;  // feature index (K dim)
          kf0.u32[i] = *(const unsigned int*)&k_lds[m * 66 + f];         // keys 0..15
          kf1.u32[i] = *(const unsigned int*)&k_lds[(16 + m) * 66 + f];  // keys 16..31
        }
        s0 = wmma_bf16(half ? qa1 : qa0, kf0, s0);
        s1 = wmma_bf16(half ? qa1 : qa0, kf1, s1);
      }

      // poly(s) = relu(a*s^2 + b*s + c), mask OOB keys, accumulate row sums,
      // and spill unnormalized weights to LDS in bf16 for re-swizzle to A-frag
#pragma unroll
      for (int i = 0; i < 8; ++i) {
        int mm = i + hi8;
        int col0 = jj + m;
        float s = s0[i] * 0.125f;  // HD^-0.5
        float p = fmaxf(ca * s * s + cb * s + cc, 0.f);
        if (col0 >= N_SEQ) p = 0.f;
        srow[i] += p;
        pw[mm * 34 + m] = f2bf(p);
        int col1 = col0 + 16;
        s = s1[i] * 0.125f;
        p = fmaxf(ca * s * s + cb * s + cc, 0.f);
        if (col1 >= N_SEQ) p = 0.f;
        srow[i] += p;
        pw[mm * 34 + 16 + m] = f2bf(p);
      }

      // attn A-frag (16x32) from per-wave LDS scratch
      Frag pa;
#pragma unroll
      for (int i = 0; i < 8; ++i) {
        int k = kbA + (i >> 2) * 16 + (i & 3) * 2;
        pa.u32[i] = *(const unsigned int*)&pw[m * 34 + k];
      }
      // oacc += attn(16x32) @ v(32x64), B-frags are aligned b32 loads
#pragma unroll
      for (int t = 0; t < 4; ++t) {
        Frag vf;
        int col = t * 16 + m;
#pragma unroll
        for (int i = 0; i < 8; ++i)
          vf.u32[i] = *(const unsigned int*)&vp_lds[(kpB + (i >> 2) * 8 + (i & 3)) * VP_STRIDE + col * 2];
        oacc[t] = wmma_bf16(pa, vf, oacc[t]);
      }
    }

    // row-sum reduction within 16-lane halves, then normalize + store
    float inv[8];
#pragma unroll
    for (int i = 0; i < 8; ++i) {
      float s = srow[i];
      s += __shfl_xor(s, 1, 32);
      s += __shfl_xor(s, 2, 32);
      s += __shfl_xor(s, 4, 32);
      s += __shfl_xor(s, 8, 32);
      inv[i] = 1.0f / (s + 1e-6f);
    }
    if (active) {
#pragma unroll
      for (int t = 0; t < 4; ++t) {
#pragma unroll
        for (int i = 0; i < 8; ++i) {
          int nrow = rt * 16 + i + hi8;
          if (nrow >= N_SEQ) continue;
          int grow = b * N_SEQ + nrow;
          out[(size_t)grow * D_MODEL + h * HD + t * 16 + m] =
              f2bf(oacc[t][i] * inv[i]);
        }
      }
    }
  }
}

// ---------------------------------------------------------------------------
// Host launcher
// ---------------------------------------------------------------------------
extern "C" void kernel_launch(void* const* d_in, const int* in_sizes, int n_in,
                              void* d_out, int out_size, void* d_ws, size_t ws_size,
                              hipStream_t stream) {
  (void)in_sizes; (void)n_in; (void)out_size; (void)ws_size;
  const float* x        = (const float*)d_in[0];
  const float* qkv_w    = (const float*)d_in[1];
  const float* qkv_b    = (const float*)d_in[2];
  const float* proj_w   = (const float*)d_in[3];
  const float* proj_b   = (const float*)d_in[4];
  const float* fc1_w    = (const float*)d_in[5];
  const float* fc1_b    = (const float*)d_in[6];
  const float* fc2_w    = (const float*)d_in[7];
  const float* fc2_b    = (const float*)d_in[8];
  const float* ln1_g    = (const float*)d_in[9];
  const float* ln1_b    = (const float*)d_in[10];
  const float* ln2_g    = (const float*)d_in[11];
  const float* ln2_b    = (const float*)d_in[12];
  const float* attn_abc = (const float*)d_in[13];
  const float* gelu_abc = (const float*)d_in[14];

  char* ws = (char*)d_ws;
  size_t off = 0;
  auto give = [&](size_t bytes) -> char* {
    char* p = ws + off;
    off += (bytes + 255) & ~(size_t)255;
    return p;
  };
  unsigned short* w_qkv  = (unsigned short*)give((size_t)384 * 1152 * 2);
  unsigned short* w_proj = (unsigned short*)give((size_t)384 * 384 * 2);
  unsigned short* w_fc1  = (unsigned short*)give((size_t)384 * 1536 * 2);
  unsigned short* w_fc2  = (unsigned short*)give((size_t)1536 * 384 * 2);
  unsigned short* h1     = (unsigned short*)give((size_t)ROWS * 384 * 2);
  unsigned short* qkv    = (unsigned short*)give((size_t)3 * QKV_STRIDE * 2);
  unsigned short* attno  = (unsigned short*)give((size_t)ROWS * 384 * 2);
  float*          x2     = (float*)give((size_t)ROWS * 384 * 4);
  unsigned short* h2     = (unsigned short*)give((size_t)ROWS * 384 * 2);
  unsigned short* h3     = (unsigned short*)give((size_t)ROWS * 1536 * 2);

  // weight conversion f32 -> bf16
  cvt_f32_bf16<<<(442368 + 255) / 256, 256, 0, stream>>>(qkv_w,  w_qkv,  442368);
  cvt_f32_bf16<<<(147456 + 255) / 256, 256, 0, stream>>>(proj_w, w_proj, 147456);
  cvt_f32_bf16<<<(589824 + 255) / 256, 256, 0, stream>>>(fc1_w,  w_fc1,  589824);
  cvt_f32_bf16<<<(589824 + 255) / 256, 256, 0, stream>>>(fc2_w,  w_fc2,  589824);

  ln_kernel<<<ROWS, 128, 0, stream>>>(x, ln1_g, ln1_b, h1);

  dim3 blk(256);
  int mby = (ROWS + 127) / 128;  // 145 row blocks
  gemm_kernel<0><<<dim3(1152 / 128, mby), blk, 0, stream>>>(
      h1, w_qkv, qkv_b, ROWS, 1152, 384, nullptr, nullptr, qkv, nullptr);

  attn_kernel<<<B_SZ * N_HEADS, blk, 0, stream>>>(
      qkv, qkv + QKV_STRIDE, qkv + 2 * (size_t)QKV_STRIDE, attn_abc, attno);

  gemm_kernel<1><<<dim3(384 / 128, mby), blk, 0, stream>>>(
      attno, w_proj, proj_b, ROWS, 384, 384, x, x2, nullptr, nullptr);

  ln_kernel<<<ROWS, 128, 0, stream>>>(x2, ln2_g, ln2_b, h2);

  gemm_kernel<2><<<dim3(1536 / 128, mby), blk, 0, stream>>>(
      h2, w_fc1, fc1_b, ROWS, 1536, 384, nullptr, nullptr, h3, gelu_abc);

  gemm_kernel<3><<<dim3(384 / 128, mby), blk, 0, stream>>>(
      h3, w_fc2, fc2_b, ROWS, 384, 1536, x2, (float*)d_out, nullptr, nullptr);
}
